// GShardRouter_35871566856543
// MI455X (gfx1250) — matrix operand order repaired and checked by reading, not compiled, not run on hardware
//
#include <hip/hip_runtime.h>
#include <hip/hip_bf16.h>

// ---------------------------------------------------------------------------
// GShard top-2 router for MI455X (gfx1250).
// B=8, S=4096, D=4096, E=64 -> 32768 tokens.
// GEMM (x @ W^T) via V_WMMA_F32_16X16X32_BF16 (f32 accumulate).
// Roofline: 17.2 GFLOP / 536 MB -> 32 FLOP/byte; at 23.3 TB/s the memory
// floor (~22us) needs ~750 TFLOP/s of matmul, i.e. the bf16 WMMA path.
// x (512 MB) is streamed once -> non-temporal b128 loads; register-staged,
// LDS double-buffered software pipeline: one barrier per K-step, global
// latency hidden behind WMMA + LDS traffic.
// ---------------------------------------------------------------------------

#define TOKENS   32768
#define DDIM     4096
#define EXP      64
#define TPB      128          // tokens per block (8 waves x 16 rows)
#define KSTEP    32
#define XT_STRIDE 36          // bf16 row stride: 72B -> 8B-aligned rows,
#define WT_STRIDE 36          // 18-bank stride (conflict-free for 16 rows)
#define LG_STRIDE 65          // padded f32 logits row
#define NBINS    4096
#define CAPACITY 1126         // int(32768 * 1.1 * 2 / 64)

#define XT_BUF   (TPB * XT_STRIDE)
#define WT_BUF   (EXP * WT_STRIDE)

typedef __bf16 bf16_t;
typedef bf16_t       v16bf __attribute__((ext_vector_type(16)));
typedef bf16_t       v2bf  __attribute__((ext_vector_type(2)));
typedef float        v8f   __attribute__((ext_vector_type(8)));
typedef float        v4f   __attribute__((ext_vector_type(4)));
typedef unsigned int v2u   __attribute__((ext_vector_type(2)));

union FragAB { unsigned int u[8]; v16bf v; };

// float4 -> 4x bf16 (RTNE fptrunc -> hardware v_cvt_pk_bf16_f32 when
// available) stored as one 8-byte LDS write (ds_store_b64).
__device__ __forceinline__ void store_bf16x4(unsigned short* lds, int hoff, v4f v) {
    union { v2bf b; unsigned int u; } lo, hi;
    lo.b.x = (bf16_t)v.x; lo.b.y = (bf16_t)v.y;
    hi.b.x = (bf16_t)v.z; hi.b.y = (bf16_t)v.w;
    v2u p; p.x = lo.u; p.y = hi.u;
    *(v2u*)(lds + hoff) = p;
}

// ---------------------------------------------------------------------------
// Kernel 0: zero dispatch/combine outputs and workspace counters.
// ---------------------------------------------------------------------------
__global__ void moe_zero_kernel(float* __restrict__ p1, long n1,
                                int* __restrict__ p2, long n2) {
    long i = (long)blockIdx.x * blockDim.x + threadIdx.x;
    long stride = (long)gridDim.x * blockDim.x;
    for (long j = i; j < n1; j += stride) p1[j] = 0.0f;
    for (long j = i; j < n2; j += stride) p2[j] = 0;
}

// ---------------------------------------------------------------------------
// Kernel 1: WMMA GEMM (logits) + softmax + top-2 + histograms.
// 256 threads = 8 wave32; wave w owns tokens [w*16, w*16+16) x all 64 experts
// as 4 accumulators of 16x16 f32.
// ---------------------------------------------------------------------------
__global__ __launch_bounds__(256) void moe_router_gemm(
    const float* __restrict__ x, const float* __restrict__ W,
    float* __restrict__ probs_out,
    int* __restrict__ top2i, float* __restrict__ top2p,
    int* __restrict__ hist, int* __restrict__ counts,
    float* __restrict__ probsum)
{
    __shared__ unsigned short Xt[2 * XT_BUF];        // 18432 B bf16 x tiles
    __shared__ unsigned short Wt[2 * WT_BUF];        //  9216 B bf16 W tiles
    __shared__ float          Lg[TPB * LG_STRIDE];   // 33280 B logits/probs
    __shared__ int            cnt[EXP];

    const int tid   = threadIdx.x;
    const int wave  = tid >> 5;
    const int lane  = tid & 31;
    const int half  = lane >> 4;      // 0: lanes 0-15, 1: lanes 16-31
    const int l15   = lane & 15;
    const int tBase = blockIdx.x * TPB;

    v8f acc[4];
    #pragma unroll
    for (int t = 0; t < 4; ++t)
        #pragma unroll
        for (int r = 0; r < 8; ++r) acc[t][r] = 0.0f;

    // per-thread staging map: thread handles float4 at (row = tid>>3 + 32*i,
    // col = (tid&7)*4) of the 128x32 x tile (i=0..3) / 64x32 W tile (i=0..1).
    const float* xp = x + (size_t)(tBase + (tid >> 3)) * DDIM + (tid & 7) * 4;
    const float* wp = W + (size_t)(tid >> 3) * DDIM + (tid & 7) * 4;
    const int xs0 = (tid >> 3) * XT_STRIDE + (tid & 7) * 4;
    const int ws0 = (tid >> 3) * WT_STRIDE + (tid & 7) * 4;

    // pipeline prologue: stage tile 0 into registers, commit to buffer 0
    v4f xr[4], wr[2];
    #pragma unroll
    for (int i = 0; i < 4; ++i)
        xr[i] = __builtin_nontemporal_load((const v4f*)(xp + (size_t)i * 32 * DDIM));
    #pragma unroll
    for (int i = 0; i < 2; ++i)
        wr[i] = *(const v4f*)(wp + (size_t)i * 32 * DDIM);
    #pragma unroll
    for (int i = 0; i < 4; ++i) store_bf16x4(Xt, xs0 + i * 32 * XT_STRIDE, xr[i]);
    #pragma unroll
    for (int i = 0; i < 2; ++i) store_bf16x4(Wt, ws0 + i * 32 * WT_STRIDE, wr[i]);
    __syncthreads();

    int buf = 0;
    for (int k0 = 0; k0 < DDIM; k0 += KSTEP) {
        const bool more = (k0 + KSTEP) < DDIM;

        // issue next tile's global loads; their wait lands at this
        // iteration's LDS commit, after the WMMA work below.
        if (more) {
            const int kn = k0 + KSTEP;
            #pragma unroll
            for (int i = 0; i < 4; ++i)
                xr[i] = __builtin_nontemporal_load(
                    (const v4f*)(xp + kn + (size_t)i * 32 * DDIM));
            #pragma unroll
            for (int i = 0; i < 2; ++i)
                wr[i] = *(const v4f*)(wp + kn + (size_t)i * 32 * DDIM);
        }

        // A fragment: 16x32 bf16 per the CDNA5 16-bit A layout.
        // half 0: v0..3 K=0..7, v4..7 K=16..23 (pairs); half 1: +8.
        const unsigned short* xb = Xt + buf * XT_BUF;
        const unsigned short* wb = Wt + buf * WT_BUF;
        FragAB A;
        const int arow = (wave * 16 + l15) * XT_STRIDE;
        #pragma unroll
        for (int i = 0; i < 8; ++i) {
            int kb = (i < 4 ? 2 * i : 16 + 2 * (i - 4)) + half * 8;
            A.u[i] = *(const unsigned int*)&xb[arow + kb];
        }

        #pragma unroll
        for (int t = 0; t < 4; ++t) {
            // B fragment: 32x16 bf16, N=lane&15; same K order as A:
            // half 0: v0..3 K=0..7, v4..7 K=16..23; half 1: +8. B[k][n] = W[n][k].
            FragAB Bf;
            const int nrow = (t * 16 + l15) * WT_STRIDE;
            #pragma unroll
            for (int v = 0; v < 8; ++v) {
                int kb = (v < 4 ? 2 * v : 16 + 2 * (v - 4)) + half * 8;
                Bf.u[v] = *(const unsigned int*)&wb[nrow + kb];
            }
            acc[t] = __builtin_amdgcn_wmma_f32_16x16x32_bf16(
                false, A.v, false, Bf.v, (short)0, acc[t], false, false);
        }

        // commit staged registers into the other buffer (its readers
        // finished last iteration), then one barrier.
        if (more) {
            unsigned short* xd = Xt + (buf ^ 1) * XT_BUF;
            unsigned short* wd = Wt + (buf ^ 1) * WT_BUF;
            #pragma unroll
            for (int i = 0; i < 4; ++i) store_bf16x4(xd, xs0 + i * 32 * XT_STRIDE, xr[i]);
            #pragma unroll
            for (int i = 0; i < 2; ++i) store_bf16x4(wd, ws0 + i * 32 * WT_STRIDE, wr[i]);
        }
        buf ^= 1;
        __syncthreads();
    }

    // spill logits to LDS. C layout: lanes 0-15 -> M=r, lanes 16-31 -> M=8+r,
    // N = lane&15.
    #pragma unroll
    for (int t = 0; t < 4; ++t)
        #pragma unroll
        for (int r = 0; r < 8; ++r) {
            int m = wave * 16 + half * 8 + r;
            int e = t * 16 + l15;
            Lg[m * LG_STRIDE + e] = acc[t][r];
        }
    if (tid < EXP) cnt[tid] = 0;
    __syncthreads();

    // per-token softmax + top-2 (threads 0..127, one token each)
    if (tid < TPB) {
        float* row = &Lg[tid * LG_STRIDE];
        float mx = row[0];
        #pragma unroll 8
        for (int e = 1; e < EXP; ++e) mx = fmaxf(mx, row[e]);
        float s = 0.0f;
        #pragma unroll 8
        for (int e = 0; e < EXP; ++e) { float p = __expf(row[e] - mx); row[e] = p; s += p; }
        float inv = 1.0f / s;
        float p1 = -1.0f; int i1 = 0;
        #pragma unroll 8
        for (int e = 0; e < EXP; ++e) {
            float p = row[e] * inv; row[e] = p;
            if (p > p1) { p1 = p; i1 = e; }     // strict '>' => lowest index on tie
        }
        float p2 = -1.0f; int i2 = 0;
        #pragma unroll 8
        for (int e = 0; e < EXP; ++e) {
            if (e != i1) { float p = row[e]; if (p > p2) { p2 = p; i2 = e; } }
        }
        float rn  = 1.0f / (p1 + p2);
        float p1n = p1 * rn, p2n = p2 * rn;
        int token = tBase + tid;
        top2i[token * 2 + 0] = i1;  top2p[token * 2 + 0] = p1n;
        top2i[token * 2 + 1] = i2;  top2p[token * 2 + 1] = p2n;
        atomicAdd(&cnt[i1], 1);
        atomicAdd(&cnt[i2], 1);
        int b1 = (int)(p1n * (float)NBINS); b1 = b1 > NBINS - 1 ? NBINS - 1 : (b1 < 0 ? 0 : b1);
        int b2 = (int)(p2n * (float)NBINS); b2 = b2 > NBINS - 1 ? NBINS - 1 : (b2 < 0 ? 0 : b2);
        atomicAdd(&hist[i1 * NBINS + b1], 1);
        atomicAdd(&hist[i2 * NBINS + b2], 1);
    }
    __syncthreads();

    // block-reduced aux-loss statistics (one global atomic per expert)
    if (tid < EXP) {
        atomicAdd(&counts[tid], cnt[tid]);
        float s = 0.0f;
        for (int m = 0; m < TPB; ++m) s += Lg[m * LG_STRIDE + tid];
        atomicAdd(&probsum[tid], s);
    }

    // coalesced probs write-out
    for (int li = tid; li < TPB * EXP; li += 256) {
        int m = li >> 6, e = li & 63;
        probs_out[(size_t)(tBase + m) * EXP + e] = Lg[m * LG_STRIDE + e];
    }
}

// ---------------------------------------------------------------------------
// Kernel 2: per-expert capacity threshold (top-down histogram scan) + aux loss
// ---------------------------------------------------------------------------
__global__ void moe_router_thresh(const int* __restrict__ hist,
                                  const int* __restrict__ counts,
                                  const float* __restrict__ probsum,
                                  int* __restrict__ thresh,
                                  int* __restrict__ remain,
                                  float* __restrict__ aux_out)
{
    __shared__ float terms[EXP];
    int e = threadIdx.x;
    if (e < EXP) {
        int cum = 0, tb = -1, rem = 0;
        for (int b = NBINS - 1; b >= 0; --b) {
            int c = hist[e * NBINS + b];
            if (cum + c >= CAPACITY) { tb = b; rem = CAPACITY - cum; break; }
            cum += c;
        }
        thresh[e] = tb;             // tb = -1 -> everything kept
        remain[e] = rem;
        terms[e] = (probsum[e] / (float)TOKENS) *
                   ((float)counts[e] / (float)(TOKENS * 2));
    }
    __syncthreads();
    if (e == 0) {
        float s = 0.0f;
        for (int i = 0; i < EXP; ++i) s += terms[i];
        *aux_out = s * (float)EXP;
    }
}

// ---------------------------------------------------------------------------
// Kernel 3: dispatch/combine scatter with capacity truncation.
// The two experts of a token are distinct, so no write conflicts per row.
// ---------------------------------------------------------------------------
__global__ void moe_router_scatter(const int* __restrict__ top2i,
                                   const float* __restrict__ top2p,
                                   const int* __restrict__ thresh,
                                   const int* __restrict__ remain,
                                   int* __restrict__ tiecnt,
                                   float* __restrict__ dispatch,
                                   float* __restrict__ combine)
{
    int s = blockIdx.x * blockDim.x + threadIdx.x;
    if (s >= TOKENS * 2) return;
    int   e = top2i[s];
    float p = top2p[s];
    int b = (int)(p * (float)NBINS); b = b > NBINS - 1 ? NBINS - 1 : (b < 0 ? 0 : b);
    int tb = thresh[e];
    bool keep = false;
    if (b > tb) keep = true;
    else if (b == tb) {
        int r = atomicAdd(&tiecnt[e], 1);
        keep = (r < remain[e]);
    }
    if (keep) {
        int token = s >> 1;
        dispatch[(size_t)token * EXP + e] = 1.0f;
        combine [(size_t)token * EXP + e] = p;
    }
}

// ---------------------------------------------------------------------------
// Workspace layout (bytes)
// ---------------------------------------------------------------------------
#define TOP2I_OFF 0u                       // 65536 ints
#define TOP2P_OFF 262144u                  // 65536 floats
#define HIST_OFF  524288u                  // 64*4096 ints
#define CNT_OFF   1572864u                 // 64 ints
#define PSUM_OFF  1573120u                 // 64 floats
#define THR_OFF   1573376u                 // 64 ints
#define REM_OFF   1573632u                 // 64 ints
#define TIE_OFF   1573888u                 // 64 ints
// total ~1.50 MB

extern "C" void kernel_launch(void* const* d_in, const int* in_sizes, int n_in,
                              void* d_out, int out_size, void* d_ws, size_t ws_size,
                              hipStream_t stream) {
    const float* x = (const float*)d_in[0];   // [8,4096,4096]
    const float* W = (const float*)d_in[1];   // [64,4096]

    float* dispatch = (float*)d_out;                          // [T,E]
    float* combine  = dispatch + (size_t)TOKENS * EXP;        // [T,E]
    float* probs    = combine  + (size_t)TOKENS * EXP;        // [T,E]
    float* aux      = probs    + (size_t)TOKENS * EXP;        // scalar

    char*  ws     = (char*)d_ws;
    int*   top2i  = (int*)  (ws + TOP2I_OFF);
    float* top2p  = (float*)(ws + TOP2P_OFF);
    int*   hist   = (int*)  (ws + HIST_OFF);
    int*   counts = (int*)  (ws + CNT_OFF);
    float* psum   = (float*)(ws + PSUM_OFF);
    int*   thr    = (int*)  (ws + THR_OFF);
    int*   rem    = (int*)  (ws + REM_OFF);
    int*   tie    = (int*)  (ws + TIE_OFF);

    // zero dispatch+combine (contiguous) and hist..tiecnt (contiguous ints)
    moe_zero_kernel<<<512, 256, 0, stream>>>(
        dispatch, (long)2 * TOKENS * EXP, hist, (long)(EXP * NBINS + 5 * EXP));

    moe_router_gemm<<<TOKENS / TPB, 256, 0, stream>>>(
        x, W, probs, top2i, top2p, hist, counts, psum);

    moe_router_thresh<<<1, 64, 0, stream>>>(hist, counts, psum, thr, rem, aux);

    moe_router_scatter<<<(TOKENS * 2) / 256, 256, 0, stream>>>(
        top2i, top2p, thr, rem, tie, dispatch, combine);
}
